// RGCN_63582695850893
// MI455X (gfx1250) — matrix operand order repaired and checked by reading, not compiled, not run on hardware
//
#include <hip/hip_runtime.h>
#include <hip/hip_bf16.h>
#include <stdint.h>

#define N_NODES 100000
#define NPAD    100096          // 782 * 128, padded row count for GEMM tiles
#define HID     128
#define NREL    8
#define NEDGE   75000
#define GRID_M  (NPAD / 128)    // 782 blocks, 8 waves x 16 rows each

typedef __bf16 bf16_t;
typedef bf16_t v16bf __attribute__((ext_vector_type(16)));
typedef bf16_t v8bf  __attribute__((ext_vector_type(8)));
typedef bf16_t v4bf  __attribute__((ext_vector_type(4)));
typedef float  v8f   __attribute__((ext_vector_type(8)));
typedef float  v4f   __attribute__((ext_vector_type(4)));

#if __has_builtin(__builtin_amdgcn_global_load_async_to_lds_b128)
#define HAVE_ASYNC_LDS 1
typedef int v4i_t __attribute__((vector_size(16)));
typedef __attribute__((address_space(1))) v4i_t glob_v4i;
typedef __attribute__((address_space(3))) v4i_t lds_v4i;
#else
#define HAVE_ASYNC_LDS 0
#endif

// ---------------------------------------------------------------------------
// Weight packing: row-major f32 W[k][n] (128x128) -> bf16 WMMA B-fragment order.
// B-matrix 32x16 bf16 layout (ISA 7.12.2 / sparse B pattern):
//   lane l holds column n = l&15; khi = l>>4; elem j -> K = j<8 ? khi*8+j
//   : 16+khi*8+(j-8), per 32-K chunk c (matches the A layout).
// Packed index = ((t*4 + c)*32 + lane)*16 + j  (t = 16-col tile, c = K chunk).
// ---------------------------------------------------------------------------
__global__ void k_pack_w(const float* __restrict__ W, bf16_t* __restrict__ out) {
  const float* Wr = W + (size_t)blockIdx.y * (HID * HID);
  bf16_t*      Or = out + (size_t)blockIdx.y * (HID * HID);
  int tid  = blockIdx.x * 256 + threadIdx.x;   // 0..16383
  int j    = tid & 15;
  int lane = (tid >> 4) & 31;
  int c    = (tid >> 9) & 3;
  int t    = (tid >> 11) & 7;
  int n    = t * 16 + (lane & 15);
  int k    = c * 32 + (j < 8 ? (lane >> 4) * 8 + j : 16 + (lane >> 4) * 8 + (j - 8));
  Or[tid] = (bf16_t)Wr[k * HID + n];
}

// ---------------------------------------------------------------------------
// Shared helpers for the 16x128 row-stripe WMMA GEMM (K = 128, bf16 x bf16 -> f32)
// ---------------------------------------------------------------------------
__device__ __forceinline__ void stage_weights(const bf16_t* __restrict__ Wp,
                                              bf16_t* ldsW) {
#if HAVE_ASYNC_LDS
  // CDNA5 async direct memory->LDS copy (GLOBAL_LOAD_ASYNC_TO_LDS_B128,
  // tracked by ASYNCcnt). 256 threads x 8 x 16B = 32KB.
  glob_v4i* g = (glob_v4i*)(uintptr_t)Wp + threadIdx.x;
  lds_v4i*  l = (lds_v4i*)(uint32_t)(uintptr_t)ldsW + threadIdx.x;
#pragma unroll
  for (int i = 0; i < 8; ++i)
    __builtin_amdgcn_global_load_async_to_lds_b128(g + i * 256, l + i * 256, 0, 0);
#if __has_builtin(__builtin_amdgcn_s_wait_asynccnt)
  __builtin_amdgcn_s_wait_asynccnt(0);
#else
  asm volatile("s_wait_asynccnt 0x0" ::: "memory");
#endif
  __syncthreads();
#else
  const uint4* s = (const uint4*)Wp;
  uint4*       d = (uint4*)ldsW;
#pragma unroll
  for (int i = 0; i < 8; ++i)
    d[threadIdx.x + i * 256] = s[threadIdx.x + i * 256];
  __syncthreads();
#endif
}

__device__ __forceinline__ v16bf load_b(const bf16_t* ldsW, int t, int c, int lane) {
  const bf16_t* p = ldsW + (((t * 4 + c) * 32 + lane) << 4);
  v8bf lo = *(const v8bf*)p;
  v8bf hi = *(const v8bf*)(p + 8);
  v16bf b;
#pragma unroll
  for (int i = 0; i < 8; ++i) { b[i] = lo[i]; b[i + 8] = hi[i]; }
  return b;
}

// A-matrix 16x32 bf16 layout (ISA 7.12.2): lane l holds row m = l&15;
// khi = l>>4; elem j -> K = c*32 + (j<8 ? khi*8 + j : 16 + khi*8 + (j-8)).
// => two contiguous 8-element (16B) loads at k-offsets c*32+khi*8 and c*32+16+khi*8.
__device__ __forceinline__ void load_a_bf16(const bf16_t* __restrict__ A, int row,
                                            int khi, v16bf a[4]) {
  const bf16_t* base = A + (size_t)row * HID + khi * 8;
#pragma unroll
  for (int c = 0; c < 4; ++c) {
    v8bf lo = *(const v8bf*)(base + c * 32);
    v8bf hi = *(const v8bf*)(base + c * 32 + 16);
    v16bf t;
#pragma unroll
    for (int i = 0; i < 8; ++i) { t[i] = lo[i]; t[i + 8] = hi[i]; }
    a[c] = t;
  }
}

__device__ __forceinline__ void load_a_f32(const float* __restrict__ A, int row,
                                           int khi, bool relu, v16bf a[4]) {
  const float* base = A + (size_t)row * HID + khi * 8;
#pragma unroll
  for (int c = 0; c < 4; ++c) {
    v4f f0 = *(const v4f*)(base + c * 32);
    v4f f1 = *(const v4f*)(base + c * 32 + 4);
    v4f f2 = *(const v4f*)(base + c * 32 + 16);
    v4f f3 = *(const v4f*)(base + c * 32 + 20);
    v16bf t;
#pragma unroll
    for (int i = 0; i < 4; ++i) {
      float a0 = f0[i], a1 = f1[i], a2 = f2[i], a3 = f3[i];
      if (relu) { a0 = fmaxf(a0, 0.f); a1 = fmaxf(a1, 0.f);
                  a2 = fmaxf(a2, 0.f); a3 = fmaxf(a3, 0.f); }
      t[i] = (bf16_t)a0; t[i + 4] = (bf16_t)a1;
      t[i + 8] = (bf16_t)a2; t[i + 12] = (bf16_t)a3;
    }
    a[c] = t;
  }
}

__device__ __forceinline__ void init_acc(v8f acc[8], const float* __restrict__ bias,
                                         int m) {
#pragma unroll
  for (int t = 0; t < 8; ++t) {
    float bv = bias ? bias[t * 16 + m] : 0.0f;
#pragma unroll
    for (int i = 0; i < 8; ++i) acc[t][i] = bv;
  }
}

// Software-pipelined 4x8 WMMA chain with one B fragment in flight: the loads
// for tile n+1 issue BEFORE the WMMA consuming tile n, so bcur/bnext live
// ranges overlap (two distinct register octets) and the pre-WMMA wait can
// relax to dscnt<=2, hiding LDS latency behind the XDL op.
__device__ __forceinline__ void mma_stripe(const v16bf a[4], const bf16_t* ldsW,
                                           int lane, v8f acc[8]) {
  v16bf bcur = load_b(ldsW, 0, 0, lane);
#pragma unroll
  for (int c = 0; c < 4; ++c) {
#pragma unroll
    for (int t = 0; t < 8; ++t) {
      v16bf bnext = bcur;
      if (!(c == 3 && t == 7)) {
        int nc = (t == 7) ? c + 1 : c;
        int nt = (t == 7) ? 0 : t + 1;
        bnext = load_b(ldsW, nt, nc, lane);
      }
      acc[t] = __builtin_amdgcn_wmma_f32_16x16x32_bf16(
          false, a[c], false, bcur, (short)0, acc[t], false, false);
      bcur = bnext;
    }
  }
}

// ---------------------------------------------------------------------------
// Layer 1: h = relu(x @ W_in + b_in), f32 input -> bf16 output (padded rows)
// ---------------------------------------------------------------------------
__global__ void k_layer1(const float* __restrict__ x, const bf16_t* __restrict__ Wp,
                         const float* __restrict__ bias, bf16_t* __restrict__ h) {
  __shared__ bf16_t ldsW[HID * HID];
  stage_weights(Wp, ldsW);
  int wave = threadIdx.x >> 5, lane = threadIdx.x & 31;
  int m = lane & 15, khi = lane >> 4;
  int row0 = (blockIdx.x * 8 + wave) * 16;
  int arow = row0 + m;
  if (arow >= N_NODES) arow = N_NODES - 1;   // clamp padded-row loads
  v16bf a[4];
  load_a_f32(x, arow, khi, false, a);
  v8f acc[8];
  init_acc(acc, bias, m);
  mma_stripe(a, ldsW, lane, acc);
#pragma unroll
  for (int t = 0; t < 8; ++t) {
    int col = t * 16 + m;
#pragma unroll
    for (int i = 0; i < 8; ++i) {
      int row = row0 + khi * 8 + i;
      float v = fmaxf(acc[t][i], 0.0f);
      h[(size_t)row * HID + col] = (bf16_t)v;
    }
  }
}

// rel_out = h @ W_self + b_self   (f32 output; seeds the atomic accumulator)
__global__ void k_gemm_f32out(const bf16_t* __restrict__ A, const bf16_t* __restrict__ Wp,
                              const float* __restrict__ bias, float* __restrict__ out) {
  __shared__ bf16_t ldsW[HID * HID];
  stage_weights(Wp, ldsW);
  int wave = threadIdx.x >> 5, lane = threadIdx.x & 31;
  int m = lane & 15, khi = lane >> 4;
  int row0 = (blockIdx.x * 8 + wave) * 16;
  v16bf a[4];
  load_a_bf16(A, row0 + m, khi, a);
  v8f acc[8];
  init_acc(acc, bias, m);
  mma_stripe(a, ldsW, lane, acc);
#pragma unroll
  for (int t = 0; t < 8; ++t) {
    int col = t * 16 + m;
#pragma unroll
    for (int i = 0; i < 8; ++i)
      out[(size_t)(row0 + khi * 8 + i) * HID + col] = acc[t][i];
  }
}

// hr = h @ W_rel_r   (bf16 output, no bias)
__global__ void k_gemm_bf16out(const bf16_t* __restrict__ A, const bf16_t* __restrict__ Wp,
                               bf16_t* __restrict__ out) {
  __shared__ bf16_t ldsW[HID * HID];
  stage_weights(Wp, ldsW);
  int wave = threadIdx.x >> 5, lane = threadIdx.x & 31;
  int m = lane & 15, khi = lane >> 4;
  int row0 = (blockIdx.x * 8 + wave) * 16;
  v16bf a[4];
  load_a_bf16(A, row0 + m, khi, a);
  v8f acc[8];
  init_acc(acc, nullptr, m);
  mma_stripe(a, ldsW, lane, acc);
#pragma unroll
  for (int t = 0; t < 8; ++t) {
    int col = t * 16 + m;
#pragma unroll
    for (int i = 0; i < 8; ++i)
      out[(size_t)(row0 + khi * 8 + i) * HID + col] = (bf16_t)acc[t][i];
  }
}

// Edge scatter: rel_out[dst] += w * hr[src]   (1 wave per edge, 4 floats/lane)
// hr (25.6MB) + rel_out (51.2MB) both L2-resident on the 192MB L2.
// Edge metadata is wave-uniform: force SGPR so src/dst/w become scalar loads.
__global__ void k_scatter(const bf16_t* __restrict__ hr, const int* __restrict__ src,
                          const int* __restrict__ dst, const float* __restrict__ w,
                          float* __restrict__ out) {
  int e = __builtin_amdgcn_readfirstlane(blockIdx.x * 8 + (threadIdx.x >> 5));
  if (e >= NEDGE) return;
  int lane = threadIdx.x & 31;
  int s = src[e], d = dst[e];
  float wt = w[e];
  v4bf v = *(const v4bf*)(hr + (size_t)s * HID + lane * 4);
  float* op = out + (size_t)d * HID + lane * 4;
  unsafeAtomicAdd(op + 0, wt * (float)v[0]);
  unsafeAtomicAdd(op + 1, wt * (float)v[1]);
  unsafeAtomicAdd(op + 2, wt * (float)v[2]);
  unsafeAtomicAdd(op + 3, wt * (float)v[3]);
}

// out = relu(rel_out) @ W_out + b_out   (ReLU fused into A-fragment load)
__global__ void k_final(const float* __restrict__ relbuf, const bf16_t* __restrict__ Wp,
                        const float* __restrict__ bias, float* __restrict__ out) {
  __shared__ bf16_t ldsW[HID * HID];
  stage_weights(Wp, ldsW);
  int wave = threadIdx.x >> 5, lane = threadIdx.x & 31;
  int m = lane & 15, khi = lane >> 4;
  int row0 = (blockIdx.x * 8 + wave) * 16;
  v16bf a[4];
  load_a_f32(relbuf, row0 + m, khi, true, a);
  v8f acc[8];
  init_acc(acc, bias, m);
  mma_stripe(a, ldsW, lane, acc);
#pragma unroll
  for (int t = 0; t < 8; ++t) {
    int col = t * 16 + m;
#pragma unroll
    for (int i = 0; i < 8; ++i) {
      int row = row0 + khi * 8 + i;
      if (row < N_NODES) out[(size_t)row * HID + col] = acc[t][i];
    }
  }
}

extern "C" void kernel_launch(void* const* d_in, const int* in_sizes, int n_in,
                              void* d_out, int out_size, void* d_ws, size_t ws_size,
                              hipStream_t stream) {
  const float* x      = (const float*)d_in[0];
  const int*   esrc   = (const int*)d_in[1];
  const int*   edst   = (const int*)d_in[2];
  const float* ew     = (const float*)d_in[3];
  const float* W_in   = (const float*)d_in[4];
  const float* b_in   = (const float*)d_in[5];
  const float* W_rel  = (const float*)d_in[6];
  const float* W_self = (const float*)d_in[7];
  const float* b_self = (const float*)d_in[8];
  const float* W_out  = (const float*)d_in[9];
  const float* b_out  = (const float*)d_in[10];
  float* out = (float*)d_out;

  char* ws = (char*)d_ws;
  size_t off = 0;
  bf16_t* h      = (bf16_t*)(ws + off); off += (size_t)NPAD * HID * 2;   // 25.6 MB
  bf16_t* hr     = (bf16_t*)(ws + off); off += (size_t)NPAD * HID * 2;   // 25.6 MB
  float*  relbuf = (float*)(ws + off);  off += (size_t)NPAD * HID * 4;   // 51.2 MB
  bf16_t* wp_in   = (bf16_t*)(ws + off); off += (size_t)HID * HID * 2;
  bf16_t* wp_self = (bf16_t*)(ws + off); off += (size_t)HID * HID * 2;
  bf16_t* wp_out  = (bf16_t*)(ws + off); off += (size_t)HID * HID * 2;
  bf16_t* wp_rel  = (bf16_t*)(ws + off); off += (size_t)NREL * HID * HID * 2;

  // 1) pack all weight matrices into bf16 WMMA fragment order
  k_pack_w<<<dim3(64, 1), 256, 0, stream>>>(W_in,   wp_in);
  k_pack_w<<<dim3(64, 1), 256, 0, stream>>>(W_self, wp_self);
  k_pack_w<<<dim3(64, 1), 256, 0, stream>>>(W_out,  wp_out);
  k_pack_w<<<dim3(64, NREL), 256, 0, stream>>>(W_rel, wp_rel);

  // 2) h = relu(x @ W_in + b_in)
  k_layer1<<<GRID_M, 256, 0, stream>>>(x, wp_in, b_in, h);

  // 3) rel_out = h @ W_self + b_self  (initializes accumulator)
  k_gemm_f32out<<<GRID_M, 256, 0, stream>>>(h, wp_self, b_self, relbuf);

  // 4) per relation: hr = h @ W_rel_r, then scatter w * hr[src] into rel_out[dst]
  for (int r = 0; r < NREL; ++r) {
    k_gemm_bf16out<<<GRID_M, 256, 0, stream>>>(h, wp_rel + (size_t)r * HID * HID, hr);
    k_scatter<<<(NEDGE + 7) / 8, 256, 0, stream>>>(
        hr, esrc + (size_t)r * NEDGE, edst + (size_t)r * NEDGE,
        ew + (size_t)r * NEDGE, relbuf);
  }

  // 5) out = relu(rel_out) @ W_out + b_out
  k_final<<<GRID_M, 256, 0, stream>>>(relbuf, wp_out, b_out, out);
}
